// QuotientGraphEncoder_84877143704149
// MI455X (gfx1250) — matrix operand order repaired and checked by reading, not compiled, not run on hardware
//
#include <hip/hip_runtime.h>
#include <stdint.h>

typedef __attribute__((ext_vector_type(16))) __bf16 v16bf;
typedef __attribute__((ext_vector_type(8)))  float  v8f;

// ---------- helpers ----------
__device__ __forceinline__ unsigned short f2bf(float f) {
  unsigned u = __float_as_uint(f);
  u = (u + 0x7FFFu + ((u >> 16) & 1u)) >> 16;   // RNE to bf16
  return (unsigned short)u;
}
__device__ __forceinline__ float siluf(float x) { return x / (1.f + __expf(-x)); }
// order-preserving float<->uint encoding for atomicMax-based segment max
__device__ __forceinline__ unsigned fenc(float f) {
  unsigned u = __float_as_uint(f);
  return (u & 0x80000000u) ? ~u : (u | 0x80000000u);
}
__device__ __forceinline__ float fdec(unsigned k) {
  unsigned u = (k & 0x80000000u) ? (k & 0x7FFFFFFFu) : ~k;
  return __uint_as_float(u);
}

// ---------- generic kernels ----------
__global__ void k_zero(float* p, long n) {
  long i = (long)blockIdx.x * blockDim.x + threadIdx.x;
  long st = (long)gridDim.x * blockDim.x;
  for (; i < n; i += st) p[i] = 0.f;
}

// w: K x N row-major fp32  ->  wt: N x K bf16 (transposed)
__global__ void k_wt(const float* __restrict__ w, unsigned short* __restrict__ wt, int K, int N) {
  long total = (long)K * N;
  for (long i = (long)blockIdx.x * blockDim.x + threadIdx.x; i < total;
       i += (long)gridDim.x * blockDim.x) {
    int k = (int)(i / N), n = (int)(i % N);
    wt[(long)n * K + k] = f2bf(w[i]);
  }
}

// C[16 x 16*NT] tile group per wave: C = Abf[M x K](bf16) * Wt^T + bias.
// Wt is N x K bf16. K compile-time & fully unrolled: per K-step the wave does
// (1+NT) x 2 b128 loads then NT back-to-back v_wmma_f32_16x16x32_bf16.
// OBF: write bf16 output (feeds another GEMM) instead of f32.
template <int K, int NT, bool OBF>
__global__ __launch_bounds__(32) void k_gemm_t(const unsigned short* __restrict__ Abf,
                                               const unsigned short* __restrict__ Wt,
                                               const float* __restrict__ bias,
                                               void* __restrict__ Cv,
                                               int M, int ldc) {
  int lane = threadIdx.x;
  int m0 = blockIdx.x * 16, n0 = blockIdx.y * (16 * NT);
  int l15 = lane & 15, hl = lane >> 4;
  int mrow = m0 + l15; if (mrow >= M) mrow = M - 1;
  const unsigned* arow = (const unsigned*)(Abf + (long)mrow * K);   // 4B-aligned
  const unsigned* wrow[NT];
#pragma unroll
  for (int t = 0; t < NT; ++t)
    wrow[t] = (const unsigned*)(Wt + (long)(n0 + 16 * t + l15) * K);
  v8f acc[NT];
#pragma unroll
  for (int t = 0; t < NT; ++t) acc[t] = (v8f){0.f,0.f,0.f,0.f,0.f,0.f,0.f,0.f};

#pragma unroll
  for (int k0 = 0; k0 < K; k0 += 32) {
    union U { unsigned u[8]; v16bf v; };
    U a, b[NT];
#pragma unroll
    for (int r = 0; r < 8; ++r) {
      // A 16x32 bf16 fragment: lanes0-15 K pairs {0..7,16..23}, lanes16-31 +8
      int ka = k0 + (r < 4 ? 2 * r : 16 + 2 * (r - 4)) + hl * 8;  // even
      a.u[r] = arow[ka >> 1];
    }
#pragma unroll
    for (int t = 0; t < NT; ++t)
#pragma unroll
      for (int r = 0; r < 8; ++r) {
        // B 32x16 fragment: lane = N, lanes0-15 K pairs {0..7,16..23}, lanes16-31 +8
        int kb = k0 + (r < 4 ? 2 * r : 16 + 2 * (r - 4)) + hl * 8;  // even
        b[t].u[r] = wrow[t][kb >> 1];
      }
#pragma unroll
    for (int t = 0; t < NT; ++t)
      acc[t] = __builtin_amdgcn_wmma_f32_16x16x32_bf16(false, a.v, false, b[t].v,
                                                       (short)0, acc[t], false, false);
  }

  // epilogue: C/D layout rows m = r + 8*hl, cols n = n0 + 16*t + l15
  float* Cf = (float*)Cv;
  unsigned short* Cb = (unsigned short*)Cv;
#pragma unroll
  for (int t = 0; t < NT; ++t) {
    int n = n0 + 16 * t + l15;
    float bb = bias ? bias[n] : 0.f;
#pragma unroll
    for (int r = 0; r < 8; ++r) {
      int m = m0 + r + 8 * hl;
      if (m < M) {
        float v = acc[t][r] + bb;
        if (OBF) Cb[(long)m * ldc + n] = f2bf(v);
        else     Cf[(long)m * ldc + n] = v;
      }
    }
  }
}

// ---------- node featurizer (outputs bf16, consumed only by GEMMs) ----------
__global__ __launch_bounds__(128) void k_node_feat(const float* __restrict__ nf,
                                                   const float* __restrict__ emb,
                                                   const float* __restrict__ c1w,
                                                   const float* __restrict__ c1b,
                                                   unsigned short* __restrict__ h1,
                                                   unsigned short* __restrict__ combin) {
  int n = blockIdx.x, t = threadIdx.x;
  __shared__ float sv[128]; __shared__ int si[128];
  float best = -3.4e38f; int bi = 0;
  for (int i = t; i < 100; i += 128) {
    float v = nf[(long)n * 103 + i];
    if (v > best) { best = v; bi = i; }
  }
  sv[t] = best; si[t] = bi; __syncthreads();
  for (int s = 64; s > 0; s >>= 1) {
    if (t < s) {
      if (sv[t + s] > sv[t] || (sv[t + s] == sv[t] && si[t + s] < si[t])) {
        sv[t] = sv[t + s]; si[t] = si[t + s];
      }
    }
    __syncthreads();
  }
  int idx = si[0];
  float f0 = nf[(long)n * 103 + 100], f1 = nf[(long)n * 103 + 101], f2 = nf[(long)n * 103 + 102];
  if (t < 64) combin[(long)n * 128 + t] = f2bf(emb[(long)idx * 64 + t]);
  for (int j = t; j < 256; j += 128) {
    float a = c1b[j] + f0 * c1w[j] + f1 * c1w[256 + j] + f2 * c1w[512 + j];
    h1[(long)n * 256 + j] = f2bf(siluf(a));
  }
}

// in-place LayerNorm(256)+SiLU; optional bf16 mirror for downstream GEMM A
__global__ __launch_bounds__(256) void k_ln_silu(float* __restrict__ buf,
                                                 const float* __restrict__ g,
                                                 const float* __restrict__ b,
                                                 unsigned short* __restrict__ mirror,
                                                 int rows) {
  int r = blockIdx.x, t = threadIdx.x;
  __shared__ float red[256];
  float v = buf[(long)r * 256 + t];
  red[t] = v; __syncthreads();
  for (int s = 128; s > 0; s >>= 1) { if (t < s) red[t] += red[t + s]; __syncthreads(); }
  float mean = red[0] / 256.f; __syncthreads();
  float d = v - mean; red[t] = d * d; __syncthreads();
  for (int s = 128; s > 0; s >>= 1) { if (t < s) red[t] += red[t + s]; __syncthreads(); }
  float var = red[0] / 256.f;
  float y = siluf(d * rsqrtf(var + 1e-5f) * g[t] + b[t]);
  buf[(long)r * 256 + t] = y;
  if (mirror) mirror[(long)r * 256 + t] = f2bf(y);
}

// edge scalar MLP stage 1 -> bf16 (consumed only by GEMM)
__global__ __launch_bounds__(128) void k_edge_mlp1(const float* __restrict__ ef, int icol, int Kin,
                                                   const float* __restrict__ w,
                                                   const float* __restrict__ b,
                                                   unsigned short* __restrict__ out) {
  int e = blockIdx.x, j = threadIdx.x;
  float acc = b[j];
  for (int i = 0; i < Kin; ++i) acc += ef[(long)e * 4 + icol + i] * w[i * 128 + j];
  out[(long)e * 128 + j] = f2bf(siluf(acc));
}

// ---------- self-loop edge_attr (segment mean) ----------
__global__ void k_cnt(const int* __restrict__ dst, int E, float* __restrict__ cnt) {
  int e = blockIdx.x * blockDim.x + threadIdx.x;
  if (e < E) atomicAdd(&cnt[dst[e]], 1.f);
}
__global__ void k_loopsum(const float* __restrict__ ea, const int* __restrict__ dst, int E,
                          float* __restrict__ loop) {
  long tid = (long)blockIdx.x * blockDim.x + threadIdx.x;
  long total = (long)E * 256;
  if (tid < total) {
    int e = (int)(tid >> 8), c = (int)(tid & 255);
    atomicAdd(&loop[(long)dst[e] * 256 + c], ea[tid]);
  }
}
__global__ void k_loopdiv(float* __restrict__ loop, const float* __restrict__ cnt,
                          unsigned short* __restrict__ mirror, int N) {
  long tid = (long)blockIdx.x * blockDim.x + threadIdx.x;
  if (tid < (long)N * 256) {
    int n = (int)(tid >> 8);
    float v = loop[tid] / fmaxf(cnt[n], 1.f);
    loop[tid] = v;
    mirror[tid] = f2bf(v);
  }
}

// ---------- GATv2 attention ----------
__global__ __launch_bounds__(32) void k_score(const float* __restrict__ xl,
                                              const float* __restrict__ xr,
                                              const float* __restrict__ ef,
                                              const int* __restrict__ src,
                                              const int* __restrict__ dst,
                                              int E, const float* __restrict__ att,
                                              int H, int shiftC,
                                              float* __restrict__ scores,
                                              unsigned* __restrict__ mx) {
  int i = blockIdx.x, lane = threadIdx.x;
  int s = (i < E) ? src[i] : (i - E);
  int d = (i < E) ? dst[i] : (i - E);
  const float* xls = xl + (long)s * 256;
  const float* xrd = xr + (long)d * 256;
  const float* er  = ef + (long)i * 256;
  float hs[4] = {0.f, 0.f, 0.f, 0.f};
#pragma unroll
  for (int j = 0; j < 8; ++j) {
    int c = lane + 32 * j;
    float m = xls[c] + xrd[c] + er[c];
    m = (m > 0.f) ? m : 0.2f * m;
    hs[c >> shiftC] += m * att[c];
  }
  for (int off = 16; off > 0; off >>= 1) {
#pragma unroll
    for (int h = 0; h < 4; ++h) hs[h] += __shfl_xor(hs[h], off, 32);
  }
  if (lane == 0) {
    for (int h = 0; h < H; ++h) {
      scores[(long)i * 4 + h] = hs[h];
      atomicMax(&mx[(long)d * 4 + h], fenc(hs[h]));
    }
  }
}

__global__ void k_expw(const float* __restrict__ scores, const unsigned* __restrict__ mx,
                       const int* __restrict__ dst, int E, int N, int H,
                       float* __restrict__ wexp, float* __restrict__ denom) {
  long tid = (long)blockIdx.x * blockDim.x + threadIdx.x;
  long total = (long)(E + N) * H;
  if (tid >= total) return;
  int i = (int)(tid / H), h = (int)(tid % H);
  int d = (i < E) ? dst[i] : (i - E);
  float w = __expf(scores[(long)i * 4 + h] - fdec(mx[(long)d * 4 + h]));
  wexp[(long)i * 4 + h] = w;
  atomicAdd(&denom[(long)d * 4 + h], w);
}

__global__ __launch_bounds__(256) void k_agg(const float* __restrict__ xl,
                                             const float* __restrict__ wexp,
                                             const float* __restrict__ denom,
                                             const int* __restrict__ src,
                                             const int* __restrict__ dst,
                                             int E, int N, int shiftC,
                                             float* __restrict__ out) {
  long tid = (long)blockIdx.x * blockDim.x + threadIdx.x;
  long total = (long)(E + N) * 256;
  if (tid >= total) return;
  int i = (int)(tid >> 8), c = (int)(tid & 255);
  int s = (i < E) ? src[i] : (i - E);
  int d = (i < E) ? dst[i] : (i - E);
  int h = c >> shiftC;
  float alpha = wexp[(long)i * 4 + h] / denom[(long)d * 4 + h];
  atomicAdd(&out[(long)d * 256 + c], alpha * xl[(long)s * 256 + c]);
}

__global__ void k_bias_relu(float* __restrict__ x, const float* __restrict__ bias,
                            unsigned short* __restrict__ mirror, int rows) {
  long tid = (long)blockIdx.x * blockDim.x + threadIdx.x;
  if (tid < (long)rows * 256) {
    int c = (int)(tid & 255);
    float v = fmaxf(x[tid] + bias[c], 0.f);
    x[tid] = v;
    mirror[tid] = f2bf(v);
  }
}

// ---------- pooling + head ----------
__global__ __launch_bounds__(256) void k_pool(const float* __restrict__ x, int rows,
                                              float* __restrict__ gacc) {
  int t = threadIdx.x;
  long r0 = (long)blockIdx.x * 256;
  float acc = 0.f;
  for (int r = 0; r < 256; ++r) {
    long rr = r0 + r;
    if (rr < rows) acc += x[rr * 256 + t];
  }
  atomicAdd(&gacc[t], acc);
}

__global__ __launch_bounds__(256) void k_head(const float* __restrict__ gsum, float nN, float nE,
                                              const float* __restrict__ flw, const float* __restrict__ flb,
                                              const float* __restrict__ flng, const float* __restrict__ flnb,
                                              const float* __restrict__ muw, const float* __restrict__ mub,
                                              const float* __restrict__ lvw, const float* __restrict__ lvb,
                                              float* __restrict__ out) {
  __shared__ float gv[512]; __shared__ float hn[256]; __shared__ float red[256];
  int t = threadIdx.x;
  gv[t] = gsum[t] / nN;
  gv[256 + t] = gsum[256 + t] / nE;
  __syncthreads();
  float acc = flb[t];
  for (int i = 0; i < 512; ++i) acc += gv[i] * flw[i * 256 + t];
  red[t] = acc; __syncthreads();
  for (int s = 128; s > 0; s >>= 1) { if (t < s) red[t] += red[t + s]; __syncthreads(); }
  float mean = red[0] / 256.f; __syncthreads();
  float d = acc - mean; red[t] = d * d; __syncthreads();
  for (int s = 128; s > 0; s >>= 1) { if (t < s) red[t] += red[t + s]; __syncthreads(); }
  float var = red[0] / 256.f;
  hn[t] = siluf(d * rsqrtf(var + 1e-5f) * flng[t] + flnb[t]);
  __syncthreads();
  if (t < 128) {
    float am = mub[t], al = lvb[t];
    for (int i = 0; i < 256; ++i) { am += hn[i] * muw[i * 128 + t]; al += hn[i] * lvw[i * 128 + t]; }
    out[t] = am;
    out[128 + t] = al;
  }
}

// ---------- host ----------
static inline int nblk(long total, int bs) { return (int)((total + bs - 1) / bs); }

// f32-output GEMMs here always have N % 64 == 0 -> NT=4
static void gemm_f32(const unsigned short* A, const unsigned short* Wt, const float* bias,
                     float* C, int M, int N, int K, int ldc, hipStream_t s) {
  dim3 g(nblk(M, 16), N / 64);
  if (K == 64)       k_gemm_t<64, 4, false><<<g, 32, 0, s>>>(A, Wt, bias, C, M, ldc);
  else if (K == 128) k_gemm_t<128, 4, false><<<g, 32, 0, s>>>(A, Wt, bias, C, M, ldc);
  else               k_gemm_t<256, 4, false><<<g, 32, 0, s>>>(A, Wt, bias, C, M, ldc);
}
// bf16-output GEMMs: (K=256,N=64) and (K=128,N=32)
static void gemm_bf16(const unsigned short* A, const unsigned short* Wt, const float* bias,
                      unsigned short* C, int M, int N, int K, int ldc, hipStream_t s) {
  if (N % 64 == 0) {
    dim3 g(nblk(M, 16), N / 64);
    k_gemm_t<256, 4, true><<<g, 32, 0, s>>>(A, Wt, bias, C, M, ldc);
  } else {
    dim3 g(nblk(M, 16), N / 32);
    k_gemm_t<128, 2, true><<<g, 32, 0, s>>>(A, Wt, bias, C, M, ldc);
  }
}

extern "C" void kernel_launch(void* const* d_in, const int* in_sizes, int n_in,
                              void* d_out, int out_size, void* d_ws, size_t ws_size,
                              hipStream_t stream) {
  (void)n_in; (void)out_size; (void)ws_size;
  const int N = in_sizes[0] / 103;
  const int E = in_sizes[1] / 2;
  const float* nf = (const float*)d_in[0];
  const int* ei = (const int*)d_in[1];
  const int* src = ei;
  const int* dst = ei + E;
  const float* ef = (const float*)d_in[2];
  // params flattened in JAX pytree (alphabetical) order after the 3 tensor inputs
  auto P = [&](int i) { return (const float*)d_in[i]; };
  struct GatP { const float *att, *bias, *llb, *llw, *lrb, *lrw, *we; int H, shift; };
  GatP gp[3] = {
    { P(3),  P(4),  P(5),  P(6),  P(7),  P(8),  P(9),  4, 6 },
    { P(10), P(11), P(12), P(13), P(14), P(15), P(16), 4, 6 },
    { P(17), P(18), P(19), P(20), P(21), P(22), P(23), 2, 7 },
  };
  const float *e_comb_b = P(24), *e_comb_w = P(25);
  const float *e_d1_b = P(26), *e_d1_w = P(27), *e_d2_b = P(28), *e_d2_w = P(29);
  const float *e_ln_b = P(30), *e_ln_g = P(31);
  const float *e_p1_b = P(32), *e_p1_w = P(33), *e_p2_b = P(34), *e_p2_w = P(35);
  const float *f_l_b = P(36), *f_l_w = P(37), *f_ln_b = P(38), *f_ln_g = P(39);
  const float *lv_b = P(40), *lv_w = P(41), *mu_b = P(42), *mu_w = P(43);
  const float *n_c1_b = P(44), *n_c1_w = P(45), *n_c2_b = P(46), *n_c2_w = P(47);
  const float *n_comb_b = P(48), *n_comb_w = P(49), *n_emb = P(50);
  const float *n_ln_b = P(51), *n_ln_g = P(52);

  float* W = (float*)d_ws;
  unsigned short* BF = (unsigned short*)d_ws;

  // ---- bf16 region (element offsets, all even) ----
  long bo = 0;
  long WT_C2 = bo;    bo += 64L * 256;
  long WT_NCOMB = bo; bo += 256L * 128;
  long WT_ECOMB = bo; bo += 256L * 64;
  long WT_D2 = bo;    bo += 32L * 128;
  long WT_P2 = bo;    bo += 32L * 128;
  long WT_CV[3][3];
  for (int c = 0; c < 3; ++c)
    for (int j = 0; j < 3; ++j) { WT_CV[c][j] = bo; bo += 256L * 256; }
  const long XBF = bo;  bo += (long)N * 256;        // bf16 mirror of node states
  const long EABF = bo; bo += (long)(E + N) * 256;  // bf16 mirror of [edge_attr;loop_attr]
  const long SBF = bo;                               // transient bf16 scratch
  {
    long ph1 = (long)N * 256 + (long)N * 128;   // h1 + comb_in
    long ph2 = (long)E * 128 + (long)E * 64;    // eh + comb_in_edge
    bo += (ph1 > ph2 ? ph1 : ph2);
  }
  const long H1B = SBF;                 // N x 256 bf16
  const long CINB = SBF + (long)N * 256; // N x 128 bf16
  const long EHB = SBF;                 // E x 128 bf16
  const long CIEB = SBF + (long)E * 128; // E x 64 bf16

  long o = (bo + 1) / 2;                // float offset after bf16 region

  // ---- f32 region ----
  const long A_EA = o; o += (long)(E + N) * 256; // [edge_attr ; loop_attr] f32
  const long B_E  = o; o += (long)(E + N) * 256; // e_full per conv
  const long XA   = o; o += (long)N * 256;
  const long XB   = o; o += (long)N * 256;
  const long XL   = o; o += (long)N * 256;
  const long XR   = o; o += (long)N * 256;
  const long SC   = o; o += (long)(E + N) * 4;
  const long WE_  = o; o += (long)(E + N) * 4;
  const long MX   = o; o += (long)N * 4;
  const long DEN  = o; o += (long)N * 4;
  const long CNT  = o; o += N;
  const long GS   = o; o += 512;

  auto zero = [&](float* p, long n) {
    int b = nblk(n, 256); if (b > 65535) b = 65535;
    k_zero<<<b, 256, 0, stream>>>(p, n);
  };
  auto cvt = [&](const float* w, long off, int K, int Nn) {
    long tot = (long)K * Nn;
    int b = nblk(tot, 256); if (b > 16384) b = 16384;
    k_wt<<<b, 256, 0, stream>>>(w, BF + off, K, Nn);
  };

  // 1) convert weights to bf16 transposed
  cvt(n_c2_w, WT_C2, 256, 64);
  cvt(n_comb_w, WT_NCOMB, 128, 256);
  cvt(e_comb_w, WT_ECOMB, 64, 256);
  cvt(e_d2_w, WT_D2, 128, 32);
  cvt(e_p2_w, WT_P2, 128, 32);
  for (int c = 0; c < 3; ++c) {
    cvt(gp[c].llw, WT_CV[c][0], 256, 256);
    cvt(gp[c].lrw, WT_CV[c][1], 256, 256);
    cvt(gp[c].we,  WT_CV[c][2], 256, 256);
  }

  // 2) node pipeline: h1(bf16) -> coord(bf16 into comb_in) -> comb GEMM -> LN+SiLU (+bf16 mirror)
  k_node_feat<<<N, 128, 0, stream>>>(nf, n_emb, n_c1_w, n_c1_b, BF + H1B, BF + CINB);
  gemm_bf16(BF + H1B, BF + WT_C2, n_c2_b, BF + CINB + 64, N, 64, 256, 128, stream);
  gemm_f32(BF + CINB, BF + WT_NCOMB, n_comb_b, W + XA, N, 256, 128, 256, stream);
  k_ln_silu<<<N, 256, 0, stream>>>(W + XA, n_ln_g, n_ln_b, BF + XBF, N);

  // 3) edge pipeline
  k_edge_mlp1<<<E, 128, 0, stream>>>(ef, 0, 1, e_d1_w, e_d1_b, BF + EHB);
  gemm_bf16(BF + EHB, BF + WT_D2, e_d2_b, BF + CIEB, E, 32, 128, 64, stream);
  k_edge_mlp1<<<E, 128, 0, stream>>>(ef, 1, 3, e_p1_w, e_p1_b, BF + EHB);
  gemm_bf16(BF + EHB, BF + WT_P2, e_p2_b, BF + CIEB + 32, E, 32, 128, 64, stream);
  gemm_f32(BF + CIEB, BF + WT_ECOMB, e_comb_b, W + A_EA, E, 256, 64, 256, stream);
  k_ln_silu<<<E, 256, 0, stream>>>(W + A_EA, e_ln_g, e_ln_b, BF + EABF, E);

  // self-loop attrs = segment-mean(edge_attr, dst) at rows E..E+N of ea_full (+bf16 mirror)
  zero(W + CNT, N);
  zero(W + A_EA + (long)E * 256, (long)N * 256);
  k_cnt<<<nblk(E, 256), 256, 0, stream>>>(dst, E, W + CNT);
  k_loopsum<<<nblk((long)E * 256, 256), 256, 0, stream>>>(W + A_EA, dst, E,
                                                          W + A_EA + (long)E * 256);
  k_loopdiv<<<nblk((long)N * 256, 256), 256, 0, stream>>>(W + A_EA + (long)E * 256, W + CNT,
                                                          BF + EABF + (long)E * 256, N);

  // 4) three GATv2 convs (ping-pong XA <-> XB; XBF always mirrors current x)
  long xcur = XA, xnext = XB;
  for (int c = 0; c < 3; ++c) {
    gemm_f32(BF + XBF, BF + WT_CV[c][0], gp[c].llb, W + XL, N, 256, 256, 256, stream);
    gemm_f32(BF + XBF, BF + WT_CV[c][1], gp[c].lrb, W + XR, N, 256, 256, 256, stream);
    gemm_f32(BF + EABF, BF + WT_CV[c][2], nullptr, W + B_E, E + N, 256, 256, 256, stream);
    zero(W + MX, (long)N * 4);
    zero(W + DEN, (long)N * 4);
    zero(W + xnext, (long)N * 256);
    k_score<<<E + N, 32, 0, stream>>>(W + XL, W + XR, W + B_E, src, dst, E,
                                      gp[c].att, gp[c].H, gp[c].shift,
                                      W + SC, (unsigned*)(W + MX));
    k_expw<<<nblk((long)(E + N) * gp[c].H, 256), 256, 0, stream>>>(
        W + SC, (const unsigned*)(W + MX), dst, E, N, gp[c].H, W + WE_, W + DEN);
    k_agg<<<nblk((long)(E + N) * 256, 256), 256, 0, stream>>>(
        W + XL, W + WE_, W + DEN, src, dst, E, N, gp[c].shift, W + xnext);
    k_bias_relu<<<nblk((long)N * 256, 256), 256, 0, stream>>>(W + xnext, gp[c].bias,
                                                              BF + XBF, N);
    long t = xcur; xcur = xnext; xnext = t;
  }

  // 5) pooling + head -> d_out (mu[0:128], log_var[128:256])
  zero(W + GS, 512);
  k_pool<<<nblk(N, 256), 256, 0, stream>>>(W + xcur, N, W + GS);
  k_pool<<<nblk(E, 256), 256, 0, stream>>>(W + A_EA, E, W + GS + 256);
  k_head<<<1, 256, 0, stream>>>(W + GS, (float)N, (float)E,
                                f_l_w, f_l_b, f_ln_g, f_ln_b,
                                mu_w, mu_b, lv_w, lv_b, (float*)d_out);
}
